// RingAttention_90615220011350
// MI455X (gfx1250) — matrix operand order repaired and checked by reading, not compiled, not run on hardware
//
#include <hip/hip_runtime.h>
#include <hip/hip_bf16.h>
#include <stdint.h>

// ---------------------------------------------------------------------------
// Problem constants (match reference)
// ---------------------------------------------------------------------------
#define BB      2
#define NN      2048
#define DD      2048
#define HEADS   16
#define KVH     4
#define DH      64
#define DINNER  1024      // HEADS*DH
#define QKVW    1536      // DINNER + 2*KVH*DH

typedef __attribute__((ext_vector_type(16))) __bf16 v16bf;
typedef __attribute__((ext_vector_type(8)))  float  v8f;

union Frag {
  v16bf v;
  uint4 q[2];
};

static __device__ __forceinline__ unsigned short f2bf(float f) {
  unsigned int u = __float_as_uint(f);
  unsigned int r = u + 0x7FFFu + ((u >> 16) & 1u);   // round-to-nearest-even
  return (unsigned short)(r >> 16);
}

static __device__ __forceinline__ v8f vzero() {
  v8f z;
#pragma unroll
  for (int i = 0; i < 8; ++i) z[i] = 0.0f;
  return z;
}

static __device__ __forceinline__ v8f wmma_bf16(const Frag& a, const Frag& b, v8f c) {
  // (neg_a, A, neg_b, B, c_mod, C, reuse_a, reuse_b)
  return __builtin_amdgcn_wmma_f32_16x16x32_bf16(false, a.v, false, b.v,
                                                 (short)0, c, false, false);
}

// ---------------------------------------------------------------------------
// Kernel 1: L2 normalize rows of x, scale by sqrt(DIM)*gamma, emit bf16
// ---------------------------------------------------------------------------
__global__ __launch_bounds__(256) void k_rmsnorm(const float* __restrict__ x,
                                                 const float* __restrict__ gamma,
                                                 unsigned short* __restrict__ xn) {
  const int row = blockIdx.x;                    // 0..4095
  const float* xr = x + (size_t)row * DD;
  float s = 0.f;
  for (int i = threadIdx.x; i < DD; i += 256) { float v = xr[i]; s += v * v; }
#pragma unroll
  for (int o = 16; o > 0; o >>= 1) s += __shfl_xor(s, o, 32);
  __shared__ float red[8];
  __shared__ float scale_s;
  if ((threadIdx.x & 31) == 0) red[threadIdx.x >> 5] = s;
  __syncthreads();
  if (threadIdx.x == 0) {
    float t = 0.f;
#pragma unroll
    for (int i = 0; i < 8; ++i) t += red[i];
    scale_s = 45.254833995939045f / fmaxf(sqrtf(t), 1e-12f);   // sqrt(2048)
  }
  __syncthreads();
  const float sc = scale_s;
  unsigned short* dst = xn + (size_t)row * DD;
  for (int i = threadIdx.x; i < DD; i += 256) dst[i] = f2bf(xr[i] * sc * gamma[i]);
}

// ---------------------------------------------------------------------------
// Kernel 2: fp32 [R x C] -> bf16 transposed [C x R]
// ---------------------------------------------------------------------------
__global__ __launch_bounds__(256) void k_convT(const float* __restrict__ W,
                                               unsigned short* __restrict__ Wt,
                                               int R, int Ccols) {
  const int idx = blockIdx.x * 256 + threadIdx.x;
  if (idx >= R * Ccols) return;
  const int r = idx / Ccols, c = idx - r * Ccols;
  Wt[(size_t)c * R + r] = f2bf(W[idx]);
}

// ---------------------------------------------------------------------------
// Kernel 3: bf16 WMMA GEMM.  C[MxN] = A[MxK] * Bt[NxK]^T, fp32 out.
// Block 128 thr = 4 waves; block tile 128x128x32; wave tile 64x64
// (4x4 WMMA accumulators -> each fragment reused 4x).
// M,N multiples of 128; K multiple of 32.
// ---------------------------------------------------------------------------
__global__ __launch_bounds__(128) void k_gemm_bf16(const unsigned short* __restrict__ A,
                                                   const unsigned short* __restrict__ Bt,
                                                   float* __restrict__ C,
                                                   int M, int N, int K) {
  __shared__ __align__(16) unsigned short sA[128 * 40];   // 32 K + 8 pad
  __shared__ __align__(16) unsigned short sB[128 * 40];
  const int tid  = threadIdx.x;
  const int lane = tid & 31;
  const int wave = tid >> 5;
  const int wr   = wave >> 1;            // 0..1
  const int wc   = wave & 1;             // 0..1
  const int l15  = lane & 15;
  const int hi   = lane >> 4;            // lane half (K sub-range select)
  const int bm   = blockIdx.y * 128;
  const int bn   = blockIdx.x * 128;

  v8f acc[4][4];
#pragma unroll
  for (int i = 0; i < 4; ++i)
#pragma unroll
    for (int j = 0; j < 2; ++j) { acc[i][2*j] = vzero(); acc[i][2*j+1] = vzero(); }

  for (int kt = 0; kt < K; kt += 32) {
    __syncthreads();
    {
      // thread t loads full 32-element row t of both tiles (4x b128 each)
      const uint4* gA = (const uint4*)(A + (size_t)(bm + tid) * K + kt);
      uint4* dA = (uint4*)&sA[tid * 40];
      dA[0] = gA[0]; dA[1] = gA[1]; dA[2] = gA[2]; dA[3] = gA[3];
      const uint4* gB = (const uint4*)(Bt + (size_t)(bn + tid) * K + kt);
      uint4* dB = (uint4*)&sB[tid * 40];
      dB[0] = gB[0]; dB[1] = gB[1]; dB[2] = gB[2]; dB[3] = gB[3];
    }
    __syncthreads();

    Frag af[4], bf[4];
#pragma unroll
    for (int i = 0; i < 4; ++i) {
      // A-layout (16-bit 16x32): lane&15 = M; lo-half K 0..7/16..23, hi-half 8..15/24..31
      const unsigned short* base = &sA[(wr * 64 + i * 16 + l15) * 40];
      af[i].q[0] = *(const uint4*)(base + (hi ? 8 : 0));
      af[i].q[1] = *(const uint4*)(base + (hi ? 24 : 16));
    }
#pragma unroll
    for (int j = 0; j < 4; ++j) {
      // B-layout (16-bit 32x16): lane&15 = N; lo-half K 0..7/16..23, hi-half 8..15/24..31
      const unsigned short* base = &sB[(wc * 64 + j * 16 + l15) * 40];
      bf[j].q[0] = *(const uint4*)(base + (hi ? 8 : 0));
      bf[j].q[1] = *(const uint4*)(base + (hi ? 24 : 16));
    }
#pragma unroll
    for (int i = 0; i < 4; ++i)
#pragma unroll
      for (int j = 0; j < 4; ++j)
        acc[i][j] = wmma_bf16(af[i], bf[j], acc[i][j]);
  }

  // C-layout: VGPR r -> row r (+8 for hi half); lane&15 = col
#pragma unroll
  for (int i = 0; i < 4; ++i)
#pragma unroll
    for (int j = 0; j < 4; ++j) {
      const int row0 = bm + wr * 64 + i * 16 + hi * 8;
      const int col  = bn + wc * 64 + j * 16 + l15;
#pragma unroll
      for (int r = 0; r < 8; ++r)
        C[(size_t)(row0 + r) * N + col] = acc[i][j][r];
    }
}

// ---------------------------------------------------------------------------
// Kernel 4: RoPE + split into head-major bf16 Q/K and d-major bf16 V^T.
// Thread handles the rotation pair (d, d+32) of one head of one token.
// ---------------------------------------------------------------------------
__global__ __launch_bounds__(256) void k_rope(const float* __restrict__ qkv,
                                              unsigned short* __restrict__ Qh,
                                              unsigned short* __restrict__ Kh,
                                              unsigned short* __restrict__ Vt) {
  const int idx = blockIdx.x * 256 + threadIdx.x;     // < 4096*24*32
  const int i   = idx & 31;
  const int hh  = (idx >> 5) % 24;                    // 0..15 q, 16..19 k, 20..23 v
  const int row = idx / (24 * 32);
  if (row >= BB * NN) return;
  const int b = row >> 11;
  const int n = row & (NN - 1);
  const float* src = qkv + (size_t)row * QKVW + hh * DH;
  float a = src[i];
  float c = src[i + 32];
  if (hh < 20) {
    // inv_freq = 10000^(-2i/64) = exp(-i * ln(1e4)/32)
    const float invf = __expf(-0.28782313662425575f * (float)i);
    const float ang  = (float)n * invf;
    float sn, cs;
    __sincosf(ang, &sn, &cs);
    const float na = a * cs - c * sn;
    const float nc = c * cs + a * sn;
    a = na; c = nc;
  }
  if (hh < 16) {
    a *= 0.125f; c *= 0.125f;                          // DIM_HEAD^-0.5
    unsigned short* dst = Qh + ((size_t)(b * HEADS + hh) * NN + n) * DH;
    dst[i] = f2bf(a); dst[i + 32] = f2bf(c);
  } else if (hh < 20) {
    unsigned short* dst = Kh + ((size_t)(b * KVH + (hh - 16)) * NN + n) * DH;
    dst[i] = f2bf(a); dst[i + 32] = f2bf(c);
  } else {
    unsigned short* dst = Vt + (size_t)(b * KVH + (hh - 20)) * DH * NN;
    dst[(size_t)i * NN + n]        = f2bf(a);
    dst[(size_t)(i + 32) * NN + n] = f2bf(c);
  }
}

// ---------------------------------------------------------------------------
// Kernel 5: causal flash attention (GQA), bf16 WMMA, fp32 online softmax.
// grid (N/64, HEADS, B), block 128 (4 waves, 16 q-rows each).
// K/V tiles staged in LDS and shared by all 4 waves; per-wave trip counts are
// identical (= qblk+1), so the block-uniform kv loop adds no extra work.
// ---------------------------------------------------------------------------
__global__ __launch_bounds__(128) void k_attn(const unsigned short* __restrict__ Qh,
                                              const unsigned short* __restrict__ Kh,
                                              const unsigned short* __restrict__ Vt,
                                              unsigned short* __restrict__ Aout) {
  __shared__ __align__(16) unsigned short sK[64 * 72];      // K tile (kv x d)
  __shared__ __align__(16) unsigned short sV[64 * 72];      // V^T tile (d x kv)
  __shared__ __align__(16) unsigned short sP[4][16 * 72];   // per-wave P tile
  const int qblk = blockIdx.x;
  const int h    = blockIdx.y;
  const int b    = blockIdx.z;
  const int kvh  = h & (KVH - 1);                           // jnp.tile -> h % KV_HEADS
  const int tid  = threadIdx.x;
  const int lane = tid & 31;
  const int wave = tid >> 5;
  const int l15  = lane & 15;
  const int hi   = lane >> 4;
  const int qr0  = qblk * 64 + wave * 16;                   // this wave's first q row

  const unsigned short* Q = Qh + ((size_t)(b * HEADS + h) * NN + qr0) * DH;
  const unsigned short* K = Kh + (size_t)(b * KVH + kvh) * NN * DH;
  const unsigned short* V = Vt + (size_t)(b * KVH + kvh) * DH * NN;

  // Q fragments for both d-chunks, kept in registers for the whole KV loop.
  Frag qf[2];
#pragma unroll
  for (int c = 0; c < 2; ++c) {
    const unsigned short* base = Q + l15 * DH + c * 32;
    qf[c].q[0] = *(const uint4*)(base + (hi ? 8 : 0));
    qf[c].q[1] = *(const uint4*)(base + (hi ? 24 : 16));
  }

  v8f o[4];
#pragma unroll
  for (int t = 0; t < 4; ++t) o[t] = vzero();
  float mrow[8], lrow[8];
#pragma unroll
  for (int r = 0; r < 8; ++r) { mrow[r] = -3.0e38f; lrow[r] = 0.f; }

  unsigned short* P = &sP[wave][0];
  const int ldRow = tid >> 1;                // 0..63
  const int ldCol = (tid & 1) * 32;          // 0 or 32
  const int nblk  = qblk + 1;                // block-uniform kv trip count

  for (int blk = 0; blk < nblk; ++blk) {
    const int kv0 = blk * 64;
    __syncthreads();                         // prior iteration done with sK/sV
    {
      const uint4* gK = (const uint4*)(K + (size_t)(kv0 + ldRow) * DH + ldCol);
      uint4* dK = (uint4*)&sK[ldRow * 72 + ldCol];
      dK[0] = gK[0]; dK[1] = gK[1]; dK[2] = gK[2]; dK[3] = gK[3];
      const uint4* gV = (const uint4*)(V + (size_t)ldRow * NN + kv0 + ldCol);
      uint4* dV = (uint4*)&sV[ldRow * 72 + ldCol];
      dV[0] = gV[0]; dV[1] = gV[1]; dV[2] = gV[2]; dV[3] = gV[3];
    }
    __syncthreads();

    // ---- S = Q K^T from the shared K tile
    v8f s[4];
#pragma unroll
    for (int t = 0; t < 4; ++t) {
      s[t] = vzero();
#pragma unroll
      for (int c = 0; c < 2; ++c) {
        Frag kf;
        const unsigned short* kb = &sK[(t * 16 + l15) * 72 + c * 32];
        kf.q[0] = *(const uint4*)(kb + (hi ? 8 : 0));
        kf.q[1] = *(const uint4*)(kb + (hi ? 24 : 16));
        s[t] = wmma_bf16(qf[c], kf, s[t]);
      }
    }
    // ---- causal mask (covers diagonal and fully-future blocks of this wave)
    if (kv0 + 63 > qr0) {
#pragma unroll
      for (int t = 0; t < 4; ++t) {
        const int col = kv0 + t * 16 + l15;
#pragma unroll
        for (int r = 0; r < 8; ++r) {
          const int row = qr0 + r + 8 * hi;
          if (col > row) s[t][r] = -3.0e38f;
        }
      }
    }
    // ---- online softmax (row stats shared across the 16-lane half)
#pragma unroll
    for (int r = 0; r < 8; ++r) {
      float mx = fmaxf(fmaxf(s[0][r], s[1][r]), fmaxf(s[2][r], s[3][r]));
#pragma unroll
      for (int off = 8; off > 0; off >>= 1) mx = fmaxf(mx, __shfl_xor(mx, off, 32));
      const float mn = fmaxf(mrow[r], mx);
      const float sc = __expf(mrow[r] - mn);
      float rs = 0.f;
#pragma unroll
      for (int t = 0; t < 4; ++t) {
        const float e = __expf(s[t][r] - mn);
        s[t][r] = e;
        rs += e;
      }
#pragma unroll
      for (int off = 8; off > 0; off >>= 1) rs += __shfl_xor(rs, off, 32);
      lrow[r] = lrow[r] * sc + rs;
      mrow[r] = mn;
#pragma unroll
      for (int t = 0; t < 4; ++t) o[t][r] *= sc;
    }
    // ---- C-layout S -> A-layout P via per-wave LDS tile (in-wave ordering)
#pragma unroll
    for (int t = 0; t < 4; ++t)
#pragma unroll
      for (int r = 0; r < 8; ++r)
        P[(r + 8 * hi) * 72 + t * 16 + l15] = f2bf(s[t][r]);
    // ---- O += P V from the shared V^T tile
#pragma unroll
    for (int c = 0; c < 2; ++c) {
      Frag pf;
      const unsigned short* pb = P + l15 * 72 + c * 32;
      pf.q[0] = *(const uint4*)(pb + (hi ? 8 : 0));
      pf.q[1] = *(const uint4*)(pb + (hi ? 24 : 16));
#pragma unroll
      for (int t = 0; t < 4; ++t) {
        Frag vf;
        const unsigned short* vb = &sV[(t * 16 + l15) * 72 + c * 32];
        vf.q[0] = *(const uint4*)(vb + (hi ? 8 : 0));
        vf.q[1] = *(const uint4*)(vb + (hi ? 24 : 16));
        o[t] = wmma_bf16(pf, vf, o[t]);
      }
    }
  }

  // ---- normalize and emit bf16 [b][n][h*64+d] for the out-projection GEMM
  unsigned short* outbase = Aout + ((size_t)b * NN + qr0) * DINNER + h * DH;
#pragma unroll
  for (int r = 0; r < 8; ++r) {
    const float inv = 1.f / lrow[r];
    const int orow = r + 8 * hi;
#pragma unroll
    for (int t = 0; t < 4; ++t)
      outbase[(size_t)orow * DINNER + t * 16 + l15] = f2bf(o[t][r] * inv);
  }
}

// ---------------------------------------------------------------------------
// Host-side orchestration
// ---------------------------------------------------------------------------
extern "C" void kernel_launch(void* const* d_in, const int* in_sizes, int n_in,
                              void* d_out, int out_size, void* d_ws, size_t ws_size,
                              hipStream_t stream) {
  const float* x     = (const float*)d_in[0];
  const float* gamma = (const float*)d_in[1];
  const float* w_qkv = (const float*)d_in[2];
  const float* w_out = (const float*)d_in[3];
  float* out = (float*)d_out;

  char* p = (char*)d_ws;
  unsigned short* xn    = (unsigned short*)p; p += (size_t)BB * NN * DD * 2;          // 16 MB
  unsigned short* wqkvT = (unsigned short*)p; p += (size_t)QKVW * DD * 2;             // 6 MB
  unsigned short* woutT = (unsigned short*)p; p += (size_t)DD * DINNER * 2;           // 4 MB
  float*          qkv   = (float*)p;          p += (size_t)BB * NN * QKVW * 4;        // 25 MB
  unsigned short* Qhd   = (unsigned short*)p; p += (size_t)BB * HEADS * NN * DH * 2;  // 8 MB
  unsigned short* Khd   = (unsigned short*)p; p += (size_t)BB * KVH * NN * DH * 2;    // 2 MB
  unsigned short* Vtd   = (unsigned short*)p; p += (size_t)BB * KVH * DH * NN * 2;    // 2 MB
  unsigned short* aout  = (unsigned short*)p; p += (size_t)BB * NN * DINNER * 2;      // 8 MB

  k_rmsnorm<<<BB * NN, 256, 0, stream>>>(x, gamma, xn);
  k_convT<<<(DD * QKVW + 255) / 256, 256, 0, stream>>>(w_qkv, wqkvT, DD, QKVW);
  k_convT<<<(DINNER * DD + 255) / 256, 256, 0, stream>>>(w_out, woutT, DINNER, DD);
  k_gemm_bf16<<<dim3(QKVW / 128, (BB * NN) / 128), 128, 0, stream>>>(
      xn, wqkvT, qkv, BB * NN, QKVW, DD);
  k_rope<<<(BB * NN * 24 * 32 + 255) / 256, 256, 0, stream>>>(qkv, Qhd, Khd, Vtd);
  k_attn<<<dim3(NN / 64, HEADS, BB), 128, 0, stream>>>(Qhd, Khd, Vtd, aout);
  k_gemm_bf16<<<dim3(DD / 128, (BB * NN) / 128), 128, 0, stream>>>(
      aout, woutT, out, BB * NN, DD, DINNER);
}
